// GNN_SEM_3143916060700
// MI455X (gfx1250) — compile-verified, matrix operand order checked by reading, not run on hardware
//
#include <hip/hip_runtime.h>
#include <hip/hip_bf16.h>

typedef __attribute__((ext_vector_type(16))) __bf16 v16bf;
typedef __attribute__((ext_vector_type(8)))  __bf16 v8bf;
typedef __attribute__((ext_vector_type(8)))  float  v8f;

#define N_NODES 100000
#define N_EDGES 400000
#define EMB 300
#define G 2048
#define N_LAYERS 5
#define L_SEM 10
#define V_SEM 30
#define NUM_TASKS 128
#define C4 (EMB / 4)   // 75 float4 chunks per 300-ch row

// ---------------------------------------------------------------------------
// Elementwise / gather / scatter kernels
// ---------------------------------------------------------------------------
__global__ void zero_f32(float* __restrict__ p, size_t n) {
  size_t i = (size_t)blockIdx.x * blockDim.x + threadIdx.x;
  if (i < n) p[i] = 0.f;
}

__global__ void copy_f32(const float* __restrict__ a, float* __restrict__ b, size_t n) {
  size_t i = (size_t)blockIdx.x * blockDim.x + threadIdx.x;
  if (i < n) b[i] = a[i];
}

__global__ void gather_emb(const int* __restrict__ idx, const float* __restrict__ emb,
                           float* __restrict__ h, size_t total) {
  size_t i = (size_t)blockIdx.x * blockDim.x + threadIdx.x;
  if (i >= total) return;
  size_t n = i / EMB; int c = (int)(i % EMB);
  h[i] = emb[(size_t)idx[n] * EMB + c];
}

// hl = h + vn[batch]
__global__ void add_vn(const float* __restrict__ h, const float* __restrict__ vn,
                       const int* __restrict__ batch, float* __restrict__ hl, size_t total) {
  size_t i = (size_t)blockIdx.x * blockDim.x + threadIdx.x;
  if (i >= total) return;
  size_t n = i / EMB; int c = (int)(i % EMB);
  hl[i] = h[i] + vn[(size_t)batch[n] * EMB + c];
}

// msg = relu(hl[src] + bond_emb[eattr]); atomic scatter onto dst
__global__ void edge_msg(const float* __restrict__ hl, const float* __restrict__ bond,
                         const int* __restrict__ src, const int* __restrict__ dst,
                         const int* __restrict__ eattr, float* __restrict__ agg, size_t total) {
  size_t i = (size_t)blockIdx.x * blockDim.x + threadIdx.x;
  if (i >= total) return;
  size_t e = i / C4; int c4 = (int)(i % C4);
  const float4 hs = *(const float4*)(hl + (size_t)src[e] * EMB + c4 * 4);
  const float4 be = *(const float4*)(bond + (size_t)eattr[e] * EMB + c4 * 4);
  float* o = agg + (size_t)dst[e] * EMB + c4 * 4;
  atomicAdd(o + 0, fmaxf(hs.x + be.x, 0.f));
  atomicAdd(o + 1, fmaxf(hs.y + be.y, 0.f));
  atomicAdd(o + 2, fmaxf(hs.z + be.z, 0.f));
  atomicAdd(o + 3, fmaxf(hs.w + be.w, 0.f));
}

// t = (1+eps[l])*hl + agg   (in place into agg)
__global__ void make_t(const float* __restrict__ hl, float* __restrict__ agg,
                       const float* __restrict__ eps, int l, size_t total) {
  size_t i = (size_t)blockIdx.x * blockDim.x + threadIdx.x;
  if (i >= total) return;
  agg[i] = (1.f + eps[l]) * hl[i] + agg[i];
}

// atomic segment-sum of [N,EMB] rows into [G,EMB] by batch id
__global__ void seg_sum(const float* __restrict__ X, const int* __restrict__ batch,
                        float* __restrict__ out, size_t total) {
  size_t i = (size_t)blockIdx.x * blockDim.x + threadIdx.x;
  if (i >= total) return;
  size_t n = i / C4; int c4 = (int)(i % C4);
  const float4 v = *(const float4*)(X + n * EMB + c4 * 4);
  float* o = out + (size_t)batch[n] * EMB + c4 * 4;
  atomicAdd(o + 0, v.x); atomicAdd(o + 1, v.y);
  atomicAdd(o + 2, v.z); atomicAdd(o + 3, v.w);
}

__global__ void count_nodes(const int* __restrict__ batch, float* __restrict__ cnt, int n) {
  int i = blockIdx.x * blockDim.x + threadIdx.x;
  if (i < n) atomicAdd(&cnt[batch[i]], 1.f);
}

__global__ void div_cnt(float* __restrict__ hg, const float* __restrict__ cnt, size_t total) {
  size_t i = (size_t)blockIdx.x * blockDim.x + threadIdx.x;
  if (i >= total) return;
  hg[i] /= fmaxf(cnt[i / EMB], 1.f);
}

// ---------------------------------------------------------------------------
// Batch norm (training mode, biased variance over rows)
// ---------------------------------------------------------------------------
__global__ void bn_stats(const float* __restrict__ X, float* __restrict__ mean,
                         float* __restrict__ var, int M, int Nc) {
  int col = blockIdx.x;
  float s = 0.f, s2 = 0.f;
  for (int r = threadIdx.x; r < M; r += blockDim.x) {
    float v = X[(size_t)r * Nc + col];
    s += v; s2 += v * v;
  }
  __shared__ float sh[256], sh2[256];
  sh[threadIdx.x] = s; sh2[threadIdx.x] = s2;
  __syncthreads();
  for (int o = blockDim.x >> 1; o > 0; o >>= 1) {
    if ((int)threadIdx.x < o) { sh[threadIdx.x] += sh[threadIdx.x + o]; sh2[threadIdx.x] += sh2[threadIdx.x + o]; }
    __syncthreads();
  }
  if (threadIdx.x == 0) {
    float m = sh[0] / (float)M;
    mean[col] = m;
    var[col] = fmaxf(sh2[0] / (float)M - m * m, 0.f);
  }
}

__global__ void bn_apply(float* __restrict__ X, const float* __restrict__ mean,
                         const float* __restrict__ var, const float* __restrict__ g,
                         const float* __restrict__ b, int Nc, int fuseRelu, size_t total) {
  size_t i = (size_t)blockIdx.x * blockDim.x + threadIdx.x;
  if (i >= total) return;
  int col = (int)(i % Nc);
  float v = (X[i] - mean[col]) * rsqrtf(var[col] + 1e-5f) * g[col] + b[col];
  X[i] = fuseRelu ? fmaxf(v, 0.f) : v;
}

// ---------------------------------------------------------------------------
// bf16 staging
// ---------------------------------------------------------------------------
// f32 [M,Nc] -> bf16 [M,Kp] (zero pad K)
__global__ void f32_to_bf16_pad(const float* __restrict__ X, __bf16* __restrict__ Y,
                                int Nc, int Kp, size_t total) {
  size_t i = (size_t)blockIdx.x * blockDim.x + threadIdx.x;
  if (i >= total) return;
  int col = (int)(i % Kp);
  size_t row = i / Kp;
  Y[i] = (col < Nc) ? (__bf16)X[row * Nc + col] : (__bf16)0.f;
}

// W f32 [K,Nout] -> Wt bf16 [Np,Kp], Wt[n][k] = W[k][n], zero pad
__global__ void w_transpose_bf16(const float* __restrict__ W, __bf16* __restrict__ Wt,
                                 int K, int Nout, int Kp, size_t total) {
  size_t i = (size_t)blockIdx.x * blockDim.x + threadIdx.x;
  if (i >= total) return;
  int k = (int)(i % Kp);
  int n = (int)(i / Kp);
  Wt[i] = (k < K && n < Nout) ? (__bf16)W[(size_t)k * Nout + n] : (__bf16)0.f;
}

// ---------------------------------------------------------------------------
// bf16 WMMA GEMM with 4x2 register blocking:
// each wave computes a 64x32 macro-tile (4 M-frags x 2 N-frags = 8 WMMA accs),
// issuing 8 v_wmma_f32_16x16x32_bf16 per 6 fragment loads (21 FLOP/B from L2
// vs 8 FLOP/B unblocked). Lane layout per CDNA5 ISA 7.12.2: lanes 0-15 hold
// K {0-7,16-23}, lanes 16-31 hold K {8-15,24-31}; A rows and pre-transposed
// W rows both load as two contiguous 16B chunks per lane.
// Requires Kp % 32 == 0, Np % 32 == 0. Ragged M handled by clamping row
// indices (loads stay in-bounds) and guarding stores.
// ---------------------------------------------------------------------------
__device__ __forceinline__ v16bf load_frag(const __bf16* p, int k) {
  v8bf lo = *(const v8bf*)(p + k);
  v8bf hi = *(const v8bf*)(p + k + 16);
  v16bf f;
#pragma unroll
  for (int i = 0; i < 8; ++i) { f[i] = lo[i]; f[i + 8] = hi[i]; }
  return f;
}

__device__ __forceinline__ void store_frag(const v8f& c, float* __restrict__ D,
                                           const float* __restrict__ bias,
                                           int rowBase, int colBase, int lane,
                                           int M, int Nout, int ldD, int fuseRelu) {
  int col = colBase + (lane & 15);
  if (col >= Nout) return;
#pragma unroll
  for (int rr = 0; rr < 8; ++rr) {
    int row = rowBase + ((lane < 16) ? rr : rr + 8);
    if (row < M) {
      float v = c[rr] + bias[col];
      if (fuseRelu) v = fmaxf(v, 0.f);
      D[(size_t)row * ldD + col] = v;
    }
  }
}

__global__ void __launch_bounds__(256)
gemm_bf16_wmma(const __bf16* __restrict__ A, const __bf16* __restrict__ Wt,
               const float* __restrict__ bias, float* __restrict__ D,
               int M, int Kp, int nMacro, int Nout, int ldD, int fuseRelu) {
  const int lane = threadIdx.x & 31;
  const int wave = threadIdx.x >> 5;
  const int tile = blockIdx.x * 8 + wave;
  const int mT = tile / nMacro;      // 64-row macro tile
  const int nT = tile % nMacro;      // 32-col macro tile
  if (mT * 64 >= M) return;

  const int half = (lane < 16) ? 0 : 8;
  const int r = lane & 15;

  // A row pointers (clamped for ragged tail; stores are guarded)
  int row0 = mT * 64 + r;
  int r0 = min(row0,      M - 1);
  int r1 = min(row0 + 16, M - 1);
  int r2 = min(row0 + 32, M - 1);
  int r3 = min(row0 + 48, M - 1);
  const __bf16* a0p = A + (size_t)r0 * Kp + half;
  const __bf16* a1p = A + (size_t)r1 * Kp + half;
  const __bf16* a2p = A + (size_t)r2 * Kp + half;
  const __bf16* a3p = A + (size_t)r3 * Kp + half;
  // W rows (Np x Kp fully allocated; no guard needed)
  const __bf16* b0p = Wt + (size_t)(nT * 32 + r)      * Kp + half;
  const __bf16* b1p = Wt + (size_t)(nT * 32 + 16 + r) * Kp + half;

  v8f c00 = {}, c01 = {}, c10 = {}, c11 = {};
  v8f c20 = {}, c21 = {}, c30 = {}, c31 = {};

  for (int k = 0; k < Kp; k += 32) {
    v16bf a0 = load_frag(a0p, k);
    v16bf a1 = load_frag(a1p, k);
    v16bf a2 = load_frag(a2p, k);
    v16bf a3 = load_frag(a3p, k);
    v16bf b0 = load_frag(b0p, k);
    v16bf b1 = load_frag(b1p, k);
    c00 = __builtin_amdgcn_wmma_f32_16x16x32_bf16(false, a0, false, b0, (short)0, c00, false, false);
    c01 = __builtin_amdgcn_wmma_f32_16x16x32_bf16(false, a0, false, b1, (short)0, c01, false, false);
    c10 = __builtin_amdgcn_wmma_f32_16x16x32_bf16(false, a1, false, b0, (short)0, c10, false, false);
    c11 = __builtin_amdgcn_wmma_f32_16x16x32_bf16(false, a1, false, b1, (short)0, c11, false, false);
    c20 = __builtin_amdgcn_wmma_f32_16x16x32_bf16(false, a2, false, b0, (short)0, c20, false, false);
    c21 = __builtin_amdgcn_wmma_f32_16x16x32_bf16(false, a2, false, b1, (short)0, c21, false, false);
    c30 = __builtin_amdgcn_wmma_f32_16x16x32_bf16(false, a3, false, b0, (short)0, c30, false, false);
    c31 = __builtin_amdgcn_wmma_f32_16x16x32_bf16(false, a3, false, b1, (short)0, c31, false, false);
  }

  const int rowB = mT * 64;
  const int colB = nT * 32;
  store_frag(c00, D, bias, rowB,      colB,      lane, M, Nout, ldD, fuseRelu);
  store_frag(c01, D, bias, rowB,      colB + 16, lane, M, Nout, ldD, fuseRelu);
  store_frag(c10, D, bias, rowB + 16, colB,      lane, M, Nout, ldD, fuseRelu);
  store_frag(c11, D, bias, rowB + 16, colB + 16, lane, M, Nout, ldD, fuseRelu);
  store_frag(c20, D, bias, rowB + 32, colB,      lane, M, Nout, ldD, fuseRelu);
  store_frag(c21, D, bias, rowB + 32, colB + 16, lane, M, Nout, ldD, fuseRelu);
  store_frag(c30, D, bias, rowB + 48, colB,      lane, M, Nout, ldD, fuseRelu);
  store_frag(c31, D, bias, rowB + 48, colB + 16, lane, M, Nout, ldD, fuseRelu);
}

// ---------------------------------------------------------------------------
// SEM head: straight-through gumbel forward == argmax one-hot -> sel_W row
// ---------------------------------------------------------------------------
__global__ void sem_head(const float* __restrict__ logits, const float* __restrict__ gumbel,
                         const float* __restrict__ selW, const float* __restrict__ selb,
                         float* __restrict__ msw, int total) {
  int i = blockIdx.x * blockDim.x + threadIdx.x;  // i = g*L_SEM + l
  if (i >= total) return;
  const float* lg = logits + (size_t)(i / L_SEM) * (L_SEM * V_SEM) + (i % L_SEM) * V_SEM;
  const float* gm = gumbel + (size_t)i * V_SEM;
  int amax = 0; float best = lg[0] + gm[0];
  for (int v = 1; v < V_SEM; ++v) {
    float y = lg[v] + gm[v];
    if (y > best) { best = y; amax = v; }
  }
  for (int j = 0; j < 3; ++j)
    msw[(size_t)i * 3 + j] = selW[amax * 3 + j] + selb[j];
}

// ---------------------------------------------------------------------------
// Host orchestration
// ---------------------------------------------------------------------------
static inline int nblk(size_t n) { return (int)((n + 255) / 256); }

static void launch_gemm(const __bf16* A, const __bf16* Wt, const float* bias, float* D,
                        int M, int Kp, int Np, int Nout, int ldD, int relu, hipStream_t s) {
  int nMacro = Np / 32;
  long tiles = (long)((M + 63) / 64) * nMacro;
  int blocks = (int)((tiles + 7) / 8);
  gemm_bf16_wmma<<<blocks, 256, 0, s>>>(A, Wt, bias, D, M, Kp, nMacro, Nout, ldD, relu);
}

static void convert_A(const float* X, __bf16* Y, int M, int Nc, int Kp, hipStream_t s) {
  size_t total = (size_t)M * Kp;
  f32_to_bf16_pad<<<nblk(total), 256, 0, s>>>(X, Y, Nc, Kp, total);
}

static void convert_W(const float* W, __bf16* Wt, int K, int Nout, int Kp, int Np, hipStream_t s) {
  size_t total = (size_t)Np * Kp;
  w_transpose_bf16<<<nblk(total), 256, 0, s>>>(W, Wt, K, Nout, Kp, total);
}

extern "C" void kernel_launch(void* const* d_in, const int* in_sizes, int n_in,
                              void* d_out, int out_size, void* d_ws, size_t ws_size,
                              hipStream_t stream) {
  (void)in_sizes; (void)n_in; (void)out_size; (void)ws_size;

  // ---- inputs (setup_inputs order) ----
  const int*   x_idx      = (const int*)d_in[0];
  const int*   e_src      = (const int*)d_in[1];
  const int*   e_dst      = e_src + N_EDGES;
  const int*   e_attr     = (const int*)d_in[2];
  const int*   batch      = (const int*)d_in[3];
  const float* gumbel     = (const float*)d_in[4];
  const float* atom_emb   = (const float*)d_in[5];
  const float* bond_emb   = (const float*)d_in[6];
  const float* eps        = (const float*)d_in[7];
  const float* gin_W1     = (const float*)d_in[8];
  const float* gin_b1     = (const float*)d_in[9];
  const float* gin_bn1_g  = (const float*)d_in[10];
  const float* gin_bn1_b  = (const float*)d_in[11];
  const float* gin_W2     = (const float*)d_in[12];
  const float* gin_b2     = (const float*)d_in[13];
  const float* node_bn_g  = (const float*)d_in[14];
  const float* node_bn_b  = (const float*)d_in[15];
  const float* vn_W1      = (const float*)d_in[16];
  const float* vn_b1      = (const float*)d_in[17];
  const float* vn_bn1_g   = (const float*)d_in[18];
  const float* vn_bn1_b   = (const float*)d_in[19];
  const float* vn_W2      = (const float*)d_in[20];
  const float* vn_b2      = (const float*)d_in[21];
  const float* vn_bn2_g   = (const float*)d_in[22];
  const float* vn_bn2_b   = (const float*)d_in[23];
  const float* pre_W      = (const float*)d_in[24];
  const float* pre_b      = (const float*)d_in[25];
  const float* sel_W      = (const float*)d_in[26];
  const float* sel_b      = (const float*)d_in[27];
  const float* aft_W      = (const float*)d_in[28];
  const float* aft_b      = (const float*)d_in[29];
  const float* hW1        = (const float*)d_in[30];
  const float* hb1        = (const float*)d_in[31];
  const float* hW2        = (const float*)d_in[32];
  const float* hb2        = (const float*)d_in[33];
  const float* hW3        = (const float*)d_in[34];
  const float* hb3        = (const float*)d_in[35];

  // ---- workspace carve (deterministic) ----
  char* ws = (char*)d_ws;
  size_t off = 0;
  auto carve = [&](size_t bytes) -> char* {
    char* p = ws + off;
    off += (bytes + 255) & ~(size_t)255;
    return p;
  };
  float*  bufA    = (float*)carve((size_t)N_NODES * EMB * 4);      // h ping
  float*  bufB    = (float*)carve((size_t)N_NODES * EMB * 4);      // h pong / agg / t
  float*  bufHL   = (float*)carve((size_t)N_NODES * EMB * 4);      // hl
  float*  bufT1   = (float*)carve((size_t)N_NODES * 600 * 4);      // hidden 600
  __bf16* bufAbf  = (__bf16*)carve((size_t)N_NODES * 640 * 2);     // bf16 A staging
  __bf16* bufWt   = (__bf16*)carve((size_t)640 * 640 * 2);         // bf16 Wt staging
  float*  bufVn   = (float*)carve((size_t)G * EMB * 4);
  float*  bufVtmp = (float*)carve((size_t)G * EMB * 4);
  float*  bufV1   = (float*)carve((size_t)G * 608 * 4);
  float*  bufMean = (float*)carve(1024 * 4);
  float*  bufVar  = (float*)carve(1024 * 4);
  float*  bufCnt  = (float*)carve((size_t)G * 4);
  float*  bufHg   = (float*)carve((size_t)G * EMB * 4);
  float*  bufLg   = (float*)carve((size_t)G * EMB * 4);
  float*  bufMsw  = (float*)carve((size_t)G * 32 * 4);
  float*  bufDis  = (float*)carve((size_t)G * EMB * 4);
  float*  bufO1   = (float*)carve((size_t)G * NUM_TASKS * 4);
  float*  bufO2   = (float*)carve((size_t)G * NUM_TASKS * 4);

  const size_t NE = (size_t)N_NODES * EMB;
  const size_t GE = (size_t)G * EMB;

  float* h = bufA;   // current node features
  float* t = bufB;   // scratch: agg -> t -> t2 -> next h

  // h = atom_emb[x_idx]; vn = 0
  gather_emb<<<nblk(NE), 256, 0, stream>>>(x_idx, atom_emb, h, NE);
  zero_f32<<<nblk(GE), 256, 0, stream>>>(bufVn, GE);

  for (int l = 0; l < N_LAYERS; ++l) {
    // hl = h + vn[batch]
    add_vn<<<nblk(NE), 256, 0, stream>>>(h, bufVn, batch, bufHL, NE);

    // agg = segment_sum(relu(hl[src] + bond_emb[l][eattr]), dst)
    zero_f32<<<nblk(NE), 256, 0, stream>>>(t, NE);
    size_t etot = (size_t)N_EDGES * C4;
    edge_msg<<<nblk(etot), 256, 0, stream>>>(bufHL, bond_emb + (size_t)l * 5 * EMB,
                                             e_src, e_dst, e_attr, t, etot);

    // t = (1+eps[l])*hl + agg
    make_t<<<nblk(NE), 256, 0, stream>>>(bufHL, t, eps, l, NE);

    // t1 = relu(bn1(t @ W1 + b1))   [N,600], Kp=320, Np=608
    convert_A(t, bufAbf, N_NODES, EMB, 320, stream);
    convert_W(gin_W1 + (size_t)l * EMB * 600, bufWt, EMB, 600, 320, 608, stream);
    launch_gemm(bufAbf, bufWt, gin_b1 + (size_t)l * 600, bufT1,
                N_NODES, 320, 608, 600, 600, 0, stream);
    bn_stats<<<600, 256, 0, stream>>>(bufT1, bufMean, bufVar, N_NODES, 600);
    bn_apply<<<nblk((size_t)N_NODES * 600), 256, 0, stream>>>(
        bufT1, bufMean, bufVar, gin_bn1_g + (size_t)l * 600, gin_bn1_b + (size_t)l * 600,
        600, 1, (size_t)N_NODES * 600);

    // t2 = node_bn(t1 @ W2 + b2) (+relu if l<4)   [N,300], Kp=608, Np=320
    convert_A(bufT1, bufAbf, N_NODES, 600, 608, stream);
    convert_W(gin_W2 + (size_t)l * 600 * EMB, bufWt, 600, EMB, 608, 320, stream);
    launch_gemm(bufAbf, bufWt, gin_b2 + (size_t)l * EMB, t,
                N_NODES, 608, 320, EMB, EMB, 0, stream);
    bn_stats<<<EMB, 256, 0, stream>>>(t, bufMean, bufVar, N_NODES, EMB);
    bn_apply<<<nblk(NE), 256, 0, stream>>>(
        t, bufMean, bufVar, node_bn_g + (size_t)l * EMB, node_bn_b + (size_t)l * EMB,
        EMB, (l < N_LAYERS - 1) ? 1 : 0, NE);

    if (l < N_LAYERS - 1) {
      // vtmp = segment_sum(hl, batch) + vn
      copy_f32<<<nblk(GE), 256, 0, stream>>>(bufVn, bufVtmp, GE);
      size_t stot = (size_t)N_NODES * C4;
      seg_sum<<<nblk(stot), 256, 0, stream>>>(bufHL, batch, bufVtmp, stot);

      // v1 = relu(bn(vtmp @ vn_W1 + b))   [G,600]
      convert_A(bufVtmp, bufAbf, G, EMB, 320, stream);
      convert_W(vn_W1 + (size_t)l * EMB * 600, bufWt, EMB, 600, 320, 608, stream);
      launch_gemm(bufAbf, bufWt, vn_b1 + (size_t)l * 600, bufV1,
                  G, 320, 608, 600, 600, 0, stream);
      bn_stats<<<600, 256, 0, stream>>>(bufV1, bufMean, bufVar, G, 600);
      bn_apply<<<nblk((size_t)G * 600), 256, 0, stream>>>(
          bufV1, bufMean, bufVar, vn_bn1_g + (size_t)l * 600, vn_bn1_b + (size_t)l * 600,
          600, 1, (size_t)G * 600);

      // vn = relu(bn(v1 @ vn_W2 + b))   [G,300]
      convert_A(bufV1, bufAbf, G, 600, 608, stream);
      convert_W(vn_W2 + (size_t)l * 600 * EMB, bufWt, 600, EMB, 608, 320, stream);
      launch_gemm(bufAbf, bufWt, vn_b2 + (size_t)l * EMB, bufVn,
                  G, 608, 320, EMB, EMB, 0, stream);
      bn_stats<<<EMB, 256, 0, stream>>>(bufVn, bufMean, bufVar, G, EMB);
      bn_apply<<<nblk(GE), 256, 0, stream>>>(
          bufVn, bufMean, bufVar, vn_bn2_g + (size_t)l * EMB, vn_bn2_b + (size_t)l * EMB,
          EMB, 1, GE);
    }

    // next layer: h <- t, scratch <- old h
    float* tmp = h; h = t; t = tmp;
  }

  // ---- global mean pool ----
  zero_f32<<<nblk((size_t)G), 256, 0, stream>>>(bufCnt, G);
  zero_f32<<<nblk(GE), 256, 0, stream>>>(bufHg, GE);
  count_nodes<<<nblk((size_t)N_NODES), 256, 0, stream>>>(batch, bufCnt, N_NODES);
  size_t stot = (size_t)N_NODES * C4;
  seg_sum<<<nblk(stot), 256, 0, stream>>>(h, batch, bufHg, stot);
  div_cnt<<<nblk(GE), 256, 0, stream>>>(bufHg, bufCnt, GE);

  // ---- SEM head ----
  // logits = hg @ pre_W + pre_b   (TAU = 1)
  convert_A(bufHg, bufAbf, G, EMB, 320, stream);
  convert_W(pre_W, bufWt, EMB, L_SEM * V_SEM, 320, 320, stream);
  launch_gemm(bufAbf, bufWt, pre_b, bufLg, G, 320, 320, L_SEM * V_SEM, L_SEM * V_SEM, 0, stream);

  // straight-through: msg_w[g,l,:] = sel_W[argmax(logits+gumbel),:] + sel_b
  sem_head<<<nblk((size_t)G * L_SEM), 256, 0, stream>>>(bufLg, gumbel, sel_W, sel_b,
                                                        bufMsw, G * L_SEM);

  // dis = msg_w @ aft_W + aft_b   [G,300], Kp=32
  convert_A(bufMsw, bufAbf, G, 3 * L_SEM, 32, stream);
  convert_W(aft_W, bufWt, 3 * L_SEM, EMB, 32, 320, stream);
  launch_gemm(bufAbf, bufWt, aft_b, bufDis, G, 32, 320, EMB, EMB, 0, stream);

  // o1 = relu(dis @ hW1 + hb1)   [G,128]
  convert_A(bufDis, bufAbf, G, EMB, 320, stream);
  convert_W(hW1, bufWt, EMB, NUM_TASKS, 320, NUM_TASKS, stream);
  launch_gemm(bufAbf, bufWt, hb1, bufO1, G, 320, NUM_TASKS, NUM_TASKS, NUM_TASKS, 1, stream);

  // o2 = relu(o1 @ hW2 + hb2)
  convert_A(bufO1, bufAbf, G, NUM_TASKS, NUM_TASKS, stream);
  convert_W(hW2, bufWt, NUM_TASKS, NUM_TASKS, NUM_TASKS, NUM_TASKS, stream);
  launch_gemm(bufAbf, bufWt, hb2, bufO2, G, NUM_TASKS, NUM_TASKS, NUM_TASKS, NUM_TASKS, 1, stream);

  // out = o2 @ hW3 + hb3  -> d_out [G, NUM_TASKS] f32
  convert_A(bufO2, bufAbf, G, NUM_TASKS, NUM_TASKS, stream);
  convert_W(hW3, bufWt, NUM_TASKS, NUM_TASKS, NUM_TASKS, NUM_TASKS, stream);
  launch_gemm(bufAbf, bufWt, hb3, (float*)d_out, G, NUM_TASKS, NUM_TASKS, NUM_TASKS,
              NUM_TASKS, 0, stream);
}
